// SimplePRMoEModel_91276644974697
// MI455X (gfx1250) — matrix operand order repaired and checked by reading, not compiled, not run on hardware
//
#include <hip/hip_runtime.h>
#include <stdint.h>

typedef __attribute__((ext_vector_type(16))) __bf16        bf16x16;
typedef __attribute__((ext_vector_type(8)))  float         f32x8;
typedef __attribute__((ext_vector_type(4)))  float         f32x4;
typedef __attribute__((ext_vector_type(4)))  unsigned int  u32x4;

#define TOK 16384
#define HD  2048
#define BM  128
#define BN  128
#define BK  32
#define LDSTR (BK + 8)   // bf16 elems per LDS row: 80 bytes (multiple of 16)

__device__ __forceinline__ unsigned short f2bf(float f) {
  unsigned int u = __builtin_bit_cast(unsigned int, f);
  u += 0x7FFFu + ((u >> 16) & 1u);          // round-to-nearest-even
  return (unsigned short)(u >> 16);
}
__device__ __forceinline__ unsigned int pack2(float a, float b) {
  return (unsigned int)f2bf(a) | ((unsigned int)f2bf(b) << 16);
}

union FragU { u32x4 u[2]; bf16x16 v; };

// ---------------------------------------------------------------------------
// Elementwise fp32 -> bf16 (for the x activation matrix)
// ---------------------------------------------------------------------------
__launch_bounds__(256)
__global__ void cvt_bf16_kernel(const float* __restrict__ src,
                                unsigned short* __restrict__ dst) {
  const size_t i = ((size_t)blockIdx.x * 256 + threadIdx.x) * 4;
  f32x4 v = *(const f32x4*)(src + i);
  unsigned int* d = (unsigned int*)(dst + i);
  d[0] = pack2(v.x, v.y);
  d[1] = pack2(v.z, v.w);
}

// ---------------------------------------------------------------------------
// Weight prep: W[e][K][N] fp32 -> Wt[e][N][K] bf16 (transposed, per expert)
// 64x64 LDS tile keeps both sides coalesced.
// ---------------------------------------------------------------------------
__launch_bounds__(256)
__global__ void wt_kernel(const float* __restrict__ W,
                          unsigned short* __restrict__ Wt) {
  __shared__ unsigned short t[64][65];
  const int e = blockIdx.z;
  const float*          We  = W  + (size_t)e * HD * HD;
  unsigned short*       Wte = Wt + (size_t)e * HD * HD;
  const int kb = blockIdx.x * 64, nb = blockIdx.y * 64;
  const int tid = threadIdx.x;
#pragma unroll
  for (int i = 0; i < 16; ++i) {
    const int lin = tid + i * 256;
    const int k = lin >> 6, n = lin & 63;
    t[k][n] = f2bf(We[(size_t)(kb + k) * HD + nb + n]);
  }
  __syncthreads();
#pragma unroll
  for (int i = 0; i < 16; ++i) {
    const int lin = tid + i * 256;
    const int n = lin >> 6, k = lin & 63;
    Wte[(size_t)(nb + n) * HD + kb + k] = t[k][n];
  }
}

// ---------------------------------------------------------------------------
// bf16 WMMA GEMM with async-LDS double-buffered staging.
// Ab: [M,K] bf16 (rows optionally gathered), Bt: [E][N][K] bf16 (transposed).
// C:  [M,N] fp32 (+bias, optionally gate-scaled/scattered); Cb: optional bf16
// mirror of C for the next layer's A operand.
// ---------------------------------------------------------------------------
template<bool GATHER>
__launch_bounds__(256)
__global__ void gemm_bf16_async(const unsigned short* __restrict__ Ab,
                                const unsigned short* __restrict__ Bt,
                                const float* __restrict__ bias,
                                float* __restrict__ C,
                                unsigned short* __restrict__ Cb,
                                const int* __restrict__ perm,
                                const int* __restrict__ off,
                                const int* __restrict__ cnt,
                                const float* __restrict__ gate) {
  __shared__ __align__(16) unsigned short As[2][BM * LDSTR];
  __shared__ __align__(16) unsigned short Bs[2][BN * LDSTR];

  const int tid  = threadIdx.x;
  const int lane = tid & 31;
  const int wave = tid >> 5;
  const int wm   = wave >> 2;     // 0..1 -> 64-row patch
  const int wn   = wave & 3;      // 0..3 -> 32-col patch
  const int half = lane >> 4;     // ISA 7.12.2 half-wave split
  const int lrow = lane & 15;
  const int mbase = blockIdx.x * BM;
  const int nbase = blockIdx.y * BN;

  int cE = TOK, oE = 0;
  const unsigned short* Be = Bt;
  const float* be = bias;
  if (GATHER) {
    const int e = blockIdx.z;
    cE = cnt[e];
    if (mbase >= cE) return;      // uniform early exit (cE>=1 beyond here)
    oE = off[e];
    Be = Bt + (size_t)e * HD * HD;
    be = bias + (size_t)e * HD;
  }

  // Each tile = 128 rows x 64B = 512 x 16B chunks; 2 chunks per thread.
  const int r0 = tid >> 2;              // 0..63
  const int c0 = (tid & 3) * 8;         // bf16 elem offset: 0,8,16,24
  const int r1 = r0 + 64;

  int tok0, tok1;
  if (GATHER) {
    const int m0 = mbase + r0, m1 = mbase + r1;
    tok0 = (m0 < cE) ? perm[oE + m0] : perm[oE];   // clamp to valid token
    tok1 = (m1 < cE) ? perm[oE + m1] : perm[oE];
  } else {
    tok0 = mbase + r0; tok1 = mbase + r1;
  }
  const unsigned short* ga0 = Ab + (size_t)tok0 * HD + c0;
  const unsigned short* ga1 = Ab + (size_t)tok1 * HD + c0;
  const unsigned short* gb0 = Be + (size_t)(nbase + r0) * HD + c0;
  const unsigned short* gb1 = Be + (size_t)(nbase + r1) * HD + c0;
  // Flat LDS pointer low 32 bits == LDS byte offset (ISA 10.2 aperture map)
  const unsigned int la0 = (unsigned int)(uintptr_t)&As[0][r0 * LDSTR + c0];
  const unsigned int la1 = (unsigned int)(uintptr_t)&As[0][r1 * LDSTR + c0];
  const unsigned int lb0 = (unsigned int)(uintptr_t)&Bs[0][r0 * LDSTR + c0];
  const unsigned int lb1 = (unsigned int)(uintptr_t)&Bs[0][r1 * LDSTR + c0];
  const unsigned int bstep = (unsigned int)(BM * LDSTR * sizeof(unsigned short));

  auto issue = [&](int buf, int k0) {
    const unsigned int d = buf ? bstep : 0u;
    asm volatile("global_load_async_to_lds_b128 %0, %1, off"
                 :: "v"(la0 + d), "v"((unsigned long long)(uintptr_t)(ga0 + k0))
                 : "memory");
    asm volatile("global_load_async_to_lds_b128 %0, %1, off"
                 :: "v"(la1 + d), "v"((unsigned long long)(uintptr_t)(ga1 + k0))
                 : "memory");
    asm volatile("global_load_async_to_lds_b128 %0, %1, off"
                 :: "v"(lb0 + d), "v"((unsigned long long)(uintptr_t)(gb0 + k0))
                 : "memory");
    asm volatile("global_load_async_to_lds_b128 %0, %1, off"
                 :: "v"(lb1 + d), "v"((unsigned long long)(uintptr_t)(gb1 + k0))
                 : "memory");
  };

  f32x8 acc[4][2] = {};
  const int NT = HD / BK;               // 64 k-steps
  issue(0, 0);
  for (int kk = 0; kk < NT; ++kk) {
    const int buf = kk & 1;
    if (kk + 1 < NT) {
      issue(buf ^ 1, (kk + 1) * BK);
      // 4 newer ops outstanding; <=4 left means current buffer landed
      asm volatile("s_wait_asynccnt 0x4" ::: "memory");
    } else {
      asm volatile("s_wait_asynccnt 0x0" ::: "memory");
    }
    __syncthreads();

    // B frags: lane<16 K0..15, lane>=16 K16..31 (N = lrow)
    FragU bfr[2];
#pragma unroll
    for (int sn = 0; sn < 2; ++sn) {
      const unsigned short* pb =
          &Bs[buf][(wn * 32 + sn * 16 + lrow) * LDSTR + half * 16];
      bfr[sn].u[0] = *(const u32x4*)(pb);
      bfr[sn].u[1] = *(const u32x4*)(pb + 8);
    }
    // A frags: lane<16 K0..15, lane>=16 K16..31 (M = lrow)
#pragma unroll
    for (int sm = 0; sm < 4; ++sm) {
      FragU af;
      const unsigned short* pa =
          &As[buf][(wm * 64 + sm * 16 + lrow) * LDSTR + half * 16];
      af.u[0] = *(const u32x4*)(pa);
      af.u[1] = *(const u32x4*)(pa + 8);
#pragma unroll
      for (int sn = 0; sn < 2; ++sn) {
        acc[sm][sn] = __builtin_amdgcn_wmma_f32_16x16x32_bf16(
            false, af.v, false, bfr[sn].v, (short)0, acc[sm][sn], false, false);
      }
    }
    __syncthreads();   // all waves done reading buf before it is re-issued
  }

  // Store: VGPR r -> (M = sm*16 + half*8 + r, N = lrow)
#pragma unroll
  for (int sm = 0; sm < 4; ++sm) {
#pragma unroll
    for (int sn = 0; sn < 2; ++sn) {
#pragma unroll
      for (int r = 0; r < 8; ++r) {
        const int mloc = wm * 64 + sm * 16 + half * 8 + r;
        const int col  = nbase + wn * 32 + sn * 16 + lrow;
        const float v  = acc[sm][sn][r];
        if (GATHER) {
          const int m = mbase + mloc;
          if (m < cE) {
            const int tok = perm[oE + m];
            const float o = gate[tok] * (v + be[col]);
            C[(size_t)tok * HD + col] = o;
            if (Cb) Cb[(size_t)tok * HD + col] = f2bf(o);
          }
        } else {
          const size_t at = (size_t)(mbase + mloc) * HD + col;
          const float o = v + be[col];
          C[at] = o;
          if (Cb) Cb[at] = f2bf(o);
        }
      }
    }
  }
}

// ---------------------------------------------------------------------------
// Top-1 gating: one wave per token. idx = argmax logits, gate = max softmax.
// ---------------------------------------------------------------------------
template<int E>
__launch_bounds__(256)
__global__ void gate_kernel(const float* __restrict__ h,
                            const float* __restrict__ wg,
                            int* __restrict__ eidx,
                            float* __restrict__ gate) {
  const int lane  = threadIdx.x & 31;
  const int token = blockIdx.x * 8 + (threadIdx.x >> 5);
  float acc[E];
#pragma unroll
  for (int e = 0; e < E; ++e) acc[e] = 0.f;
  const float* hr = h + (size_t)token * HD;
  for (int k = lane; k < HD; k += 32) {
    const float xv = hr[k];
    const float* w = wg + (size_t)k * E;
#pragma unroll
    for (int e = 0; e < E; ++e) acc[e] += xv * w[e];
  }
#pragma unroll
  for (int e = 0; e < E; ++e) {
    acc[e] += __shfl_xor(acc[e], 16);
    acc[e] += __shfl_xor(acc[e], 8);
    acc[e] += __shfl_xor(acc[e], 4);
    acc[e] += __shfl_xor(acc[e], 2);
    acc[e] += __shfl_xor(acc[e], 1);
  }
  if (lane == 0) {
    float mx = acc[0]; int am = 0;
#pragma unroll
    for (int e = 1; e < E; ++e) if (acc[e] > mx) { mx = acc[e]; am = e; }
    float s = 0.f;
#pragma unroll
    for (int e = 0; e < E; ++e) s += __expf(acc[e] - mx);
    eidx[token] = am;
    gate[token] = 1.f / s;        // softmax value at the argmax
  }
}

// ---------------------------------------------------------------------------
// Routing helpers
// ---------------------------------------------------------------------------
__global__ void zero_kernel(int* p, int n) {
  const int i = blockIdx.x * 256 + threadIdx.x;
  if (i < n) p[i] = 0;
}
__global__ void count_kernel(const int* __restrict__ eidx, int* __restrict__ cnt) {
  const int t = blockIdx.x * 256 + threadIdx.x;
  if (t < TOK) atomicAdd(&cnt[eidx[t]], 1);
}
__global__ void offsets_kernel(const int* __restrict__ cnt, int* __restrict__ off,
                               int* __restrict__ run, int E) {
  if (threadIdx.x == 0 && blockIdx.x == 0) {
    int a = 0;
    for (int e = 0; e < E; ++e) { off[e] = a; run[e] = a; a += cnt[e]; }
  }
}
__global__ void scatter_kernel(const int* __restrict__ eidx, int* __restrict__ run,
                               int* __restrict__ perm) {
  const int t = blockIdx.x * 256 + threadIdx.x;
  if (t < TOK) { const int p = atomicAdd(&run[eidx[t]], 1); perm[p] = t; }
}

// ---------------------------------------------------------------------------
// emb[b,c] = mean_s (h + o)   (residual folded in)
// ---------------------------------------------------------------------------
__launch_bounds__(256)
__global__ void emb_kernel(const float* __restrict__ h, const float* __restrict__ o,
                           float* __restrict__ emb) {
  const int col = blockIdx.x * 256 + threadIdx.x;
  const int b   = blockIdx.y;
  const float* hp = h + (size_t)b * 2048 * HD + col;
  const float* op = o + (size_t)b * 2048 * HD + col;
  float s = 0.f;
  for (int t = 0; t < 2048; ++t) s += hp[(size_t)t * HD] + op[(size_t)t * HD];
  emb[b * HD + col] = s * (1.0f / 2048.0f);
}

// ---------------------------------------------------------------------------
// loss = mean_b (logsumexp(emb[b]) - emb[b, y[b]])
// ---------------------------------------------------------------------------
__launch_bounds__(256)
__global__ void loss_kernel(const float* __restrict__ emb, const int* __restrict__ y,
                            float* __restrict__ out) {
  __shared__ float red[256];
  const int tid = threadIdx.x;
  float loss = 0.f;
  for (int b = 0; b < 8; ++b) {
    const float* eb = emb + (size_t)b * HD;
    float m = -3.4e38f;
    for (int c = tid; c < HD; c += 256) m = fmaxf(m, eb[c]);
    red[tid] = m; __syncthreads();
    for (int s = 128; s > 0; s >>= 1) {
      if (tid < s) red[tid] = fmaxf(red[tid], red[tid + s]);
      __syncthreads();
    }
    const float mx = red[0]; __syncthreads();
    float se = 0.f;
    for (int c = tid; c < HD; c += 256) se += __expf(eb[c] - mx);
    red[tid] = se; __syncthreads();
    for (int s = 128; s > 0; s >>= 1) {
      if (tid < s) red[tid] += red[tid + s];
      __syncthreads();
    }
    if (tid == 0) loss += mx + __logf(red[0]) - eb[y[b]];
    __syncthreads();
  }
  if (tid == 0) out[0] = loss * 0.125f;
}

// ---------------------------------------------------------------------------
extern "C" void kernel_launch(void* const* d_in, const int* in_sizes, int n_in,
                              void* d_out, int out_size, void* d_ws, size_t ws_size,
                              hipStream_t stream) {
  (void)in_sizes; (void)n_in; (void)out_size; (void)ws_size;
  const float* x   = (const float*)d_in[0];
  const int*   y   = (const int*)  d_in[1];
  const float* w1  = (const float*)d_in[2];
  const float* b1  = (const float*)d_in[3];
  const float* wg2 = (const float*)d_in[4];
  const float* W2  = (const float*)d_in[5];
  const float* b2  = (const float*)d_in[6];
  const float* wg3 = (const float*)d_in[7];
  const float* W3  = (const float*)d_in[8];
  const float* b3  = (const float*)d_in[9];

  char* p = (char*)d_ws;
  const size_t fbytes = (size_t)TOK * HD * sizeof(float);           // 134 MB
  const size_t bbytes = (size_t)TOK * HD * sizeof(unsigned short);  // 67 MB
  float* h    = (float*)p;           p += fbytes;
  float* o1   = (float*)p;           p += fbytes;
  float* o2   = (float*)p;           p += fbytes;
  unsigned short* xb  = (unsigned short*)p; p += bbytes;
  unsigned short* hb  = (unsigned short*)p; p += bbytes;
  unsigned short* o1b = (unsigned short*)p; p += bbytes;
  unsigned short* w1t = (unsigned short*)p; p += (size_t)1  * HD * HD * 2;
  unsigned short* W2t = (unsigned short*)p; p += (size_t)8  * HD * HD * 2;
  unsigned short* W3t = (unsigned short*)p; p += (size_t)16 * HD * HD * 2;
  float* emb  = (float*)p;           p += (size_t)8 * HD * sizeof(float);
  int*   idx2 = (int*)  p;           p += (size_t)TOK * 4;
  int*   perm2= (int*)  p;           p += (size_t)TOK * 4;
  float* gate2= (float*)p;           p += (size_t)TOK * 4;
  int*   idx3 = (int*)  p;           p += (size_t)TOK * 4;
  int*   perm3= (int*)  p;           p += (size_t)TOK * 4;
  float* gate3= (float*)p;           p += (size_t)TOK * 4;
  int*   cnt2 = (int*)  p;           p += 16 * 4;
  int*   cnt3 = (int*)  p;           p += 16 * 4;   // contiguous with cnt2
  int*   off2 = (int*)  p;           p += 16 * 4;
  int*   run2 = (int*)  p;           p += 16 * 4;
  int*   off3 = (int*)  p;           p += 16 * 4;
  int*   run3 = (int*)  p;           p += 16 * 4;

  const dim3 blk(256);

  zero_kernel<<<1, 64, 0, stream>>>(cnt2, 32);  // cnt2 + cnt3

  // Operand prep: bf16 activations + transposed bf16 weights
  cvt_bf16_kernel<<<(TOK * HD) / 1024, blk, 0, stream>>>(x, xb);
  wt_kernel<<<dim3(HD / 64, HD / 64, 1),  blk, 0, stream>>>(w1, w1t);
  wt_kernel<<<dim3(HD / 64, HD / 64, 8),  blk, 0, stream>>>(W2, W2t);
  wt_kernel<<<dim3(HD / 64, HD / 64, 16), blk, 0, stream>>>(W3, W3t);

  // 1) h = x @ w1 + b1   (also emits bf16 mirror hb)
  gemm_bf16_async<false><<<dim3(TOK / BM, HD / BN), blk, 0, stream>>>(
      xb, w1t, b1, h, hb, nullptr, nullptr, nullptr, nullptr);

  // 2) MoE layer 2 (E=8)
  gate_kernel<8><<<TOK / 8, blk, 0, stream>>>(h, wg2, idx2, gate2);
  count_kernel<<<TOK / 256, blk, 0, stream>>>(idx2, cnt2);
  offsets_kernel<<<1, 1, 0, stream>>>(cnt2, off2, run2, 8);
  scatter_kernel<<<TOK / 256, blk, 0, stream>>>(idx2, run2, perm2);
  gemm_bf16_async<true><<<dim3(TOK / BM, HD / BN, 8), blk, 0, stream>>>(
      hb, W2t, b2, o1, o1b, perm2, off2, cnt2, gate2);

  // 3) MoE layer 3 (E=16)
  gate_kernel<16><<<TOK / 8, blk, 0, stream>>>(o1, wg3, idx3, gate3);
  count_kernel<<<TOK / 256, blk, 0, stream>>>(idx3, cnt3);
  offsets_kernel<<<1, 1, 0, stream>>>(cnt3, off3, run3, 16);
  scatter_kernel<<<TOK / 256, blk, 0, stream>>>(idx3, run3, perm3);
  gemm_bf16_async<true><<<dim3(TOK / BM, HD / BN, 16), blk, 0, stream>>>(
      o1b, W3t, b3, o2, nullptr, perm3, off3, cnt3, gate3);

  // 4) residual + sequence mean, then log-softmax NLL loss
  emb_kernel<<<dim3(HD / 256, 8), blk, 0, stream>>>(h, o2, emb);
  loss_kernel<<<1, blk, 0, stream>>>(emb, y, (float*)d_out);
}
